// GridNet3D_23862838297138
// MI455X (gfx1250) — matrix operand order repaired and checked by reading, not compiled, not run on hardware
//
#include <hip/hip_runtime.h>
#include <hip/hip_bf16.h>

typedef __attribute__((ext_vector_type(16))) _Float16 v16h;
typedef __attribute__((ext_vector_type(2)))  __fp16   fp16x2;
typedef __attribute__((ext_vector_type(8)))  float    v8f;
typedef __attribute__((ext_vector_type(4)))  float    v4f;
typedef __attribute__((ext_vector_type(2)))  float    v2f;

#define NGRID   128
#define FEAT    16
#define WIDTH   64
#define NPOSENC 9          // 3 freqs * 3 dims
#define KPAD    32         // 25 inputs padded to 32 for WMMA K
#define BLOCK   256        // 8 waves of 32

union V16U {
    uint4 q[2];
    v16h  h;
};

__device__ __forceinline__ unsigned int pk2(float a, float b)
{
    // v_cvt_pk_rtz_f16_f32: two f32 -> packed f16 in one DWORD
    union { fp16x2 h; unsigned int u; } c;
    c.h = __builtin_amdgcn_cvt_pkrtz(a, b);
    return c.u;
}

// XOR-butterfly lane exchange via ds_swizzle_b32 group-of-32 mode
// (offset[14:10]=xor_mask, [9:5]=or_mask=0, [4:0]=and_mask=0x1F)
template <int PAT>
__device__ __forceinline__ float swz_xor(float v)
{
    union { float f; int i; } u;
    u.f = v;
    u.i = __builtin_amdgcn_ds_swizzle(u.i, PAT);
    return u.f;
}

template <int XORMASK>
__device__ __forceinline__ void butterfly_step(float* psum)
{
    constexpr int PAT = (XORMASK << 10) | 0x1F;
    #pragma unroll
    for (int r = 0; r < 8; ++r)
        psum[r] += swz_xor<PAT>(psum[r]);
}

__device__ __forceinline__ void axis_setup(float x, float& t_out, float* w, int* idx)
{
    float u  = x * (float)(NGRID - 1);
    float fi = floorf(u);
    float t  = u - fi;
    int   i  = (int)fi;
    float t2 = t * t;
    float t3 = t2 * t;
    // Catmull-Rom cubic weights (matches reference)
    w[0] = 0.5f * (-t3 + 2.0f * t2 - t);
    w[1] = 0.5f * ( 3.0f * t3 - 5.0f * t2 + 2.0f);
    w[2] = 0.5f * (-3.0f * t3 + 4.0f * t2 + t);
    w[3] = 0.5f * ( t3 - t2);
    idx[0] = min(max(i - 1, 0), NGRID - 1);
    idx[1] = min(max(i    , 0), NGRID - 1);
    idx[2] = min(max(i + 1, 0), NGRID - 1);
    idx[3] = min(max(i + 2, 0), NGRID - 1);
    t_out = t;
}

__launch_bounds__(BLOCK, 4)
__global__ void gridnet3d_kernel(const float* __restrict__ x,
                                 const float* __restrict__ grid,
                                 const float* __restrict__ w1,
                                 const float* __restrict__ b1,
                                 const float* __restrict__ w2,
                                 const float* __restrict__ b2,
                                 float* __restrict__ out,
                                 int npts)
{
    // LDS: f16 W1 [64][32] (B-operand source) + per-wave f16 activation rows
    __shared__ __align__(16) _Float16 sW1[WIDTH * KPAD];        // 4 KB
    __shared__ __align__(16) _Float16 sH[8 * 32 * KPAD];        // 16 KB (8 waves x 32 rows x 32 halfs)

    const int tid  = threadIdx.x;
    const int wv   = tid >> 5;          // wave id in block (0..7)
    const int lane = tid & 31;

    // ---- Stage w1 -> f16 in LDS, padded K 25 -> 32 ----
    #pragma unroll
    for (int i = tid; i < WIDTH * KPAD; i += BLOCK) {
        int j = i >> 5;      // hidden unit
        int k = i & 31;      // input channel
        sW1[i] = (k < 25) ? (_Float16)w1[j * 25 + k] : (_Float16)0.0f;
    }

    // ---- Per-lane point: tricubic gather + positional encodings ----
    const int p    = blockIdx.x * BLOCK + tid;
    const int p_ld = min(p, npts - 1);

    float x0 = x[3 * p_ld + 0];
    float x1 = x[3 * p_ld + 1];
    float x2 = x[3 * p_ld + 2];

    float tx, ty, tz;
    float wa[4], wb[4], wc[4];
    int   ia[4], ib[4], ic[4];
    axis_setup(x0, tx, wa, ia);
    axis_setup(x1, ty, wb, ib);
    axis_setup(x2, tz, wc, ic);

    // 16 feature accumulators as 8 x <2 x float>; backend packs/dual-issues
    const v4f* gv = (const v4f*)grid;   // grid cell = 4 x float4 (64 B contiguous)
    v2f acc[8];
    #pragma unroll
    for (int i = 0; i < 8; ++i) acc[i] = (v2f){0.f, 0.f};

    #pragma unroll
    for (int a = 0; a < 4; ++a) {
        const int ra = ia[a] * NGRID;
        #pragma unroll
        for (int b = 0; b < 4; ++b) {
            const float wab = wa[a] * wb[b];
            const int   rb  = (ra + ib[b]) * NGRID;
            #pragma unroll
            for (int c = 0; c < 4; ++c) {
                const float w  = wab * wc[c];
                const v2f   wv2 = (v2f){w, w};
                const v4f*  gp = gv + ((rb + ic[c]) << 2);  // cell base in float4 units
                v4f g0 = gp[0], g1 = gp[1], g2 = gp[2], g3 = gp[3];
                acc[0] = __builtin_elementwise_fma(wv2, __builtin_shufflevector(g0, g0, 0, 1), acc[0]);
                acc[1] = __builtin_elementwise_fma(wv2, __builtin_shufflevector(g0, g0, 2, 3), acc[1]);
                acc[2] = __builtin_elementwise_fma(wv2, __builtin_shufflevector(g1, g1, 0, 1), acc[2]);
                acc[3] = __builtin_elementwise_fma(wv2, __builtin_shufflevector(g1, g1, 2, 3), acc[3]);
                acc[4] = __builtin_elementwise_fma(wv2, __builtin_shufflevector(g2, g2, 0, 1), acc[4]);
                acc[5] = __builtin_elementwise_fma(wv2, __builtin_shufflevector(g2, g2, 2, 3), acc[5]);
                acc[6] = __builtin_elementwise_fma(wv2, __builtin_shufflevector(g3, g3, 0, 1), acc[6]);
                acc[7] = __builtin_elementwise_fma(wv2, __builtin_shufflevector(g3, g3, 2, 3), acc[7]);
            }
        }
    }

    // sin positional encodings: order [k=1:(x,y,z), k=2:(x,y,z), k=3:(x,y,z)]
    const float TWO_PI = 6.283185307179586f;
    float pos[NPOSENC];
    #pragma unroll
    for (int k = 0; k < 3; ++k) {
        float m = TWO_PI * (float)(k + 1);
        pos[k * 3 + 0] = __sinf(m * tx);
        pos[k * 3 + 1] = __sinf(m * ty);
        pos[k * 3 + 2] = __sinf(m * tz);
    }

    // ---- Pack h row (f16, K=32) via v_cvt_pk_rtz_f16_f32, store to LDS ----
    uint4 q0, q1, q2, q3;
    q0.x = pk2(acc[0][0], acc[0][1]);  q0.y = pk2(acc[1][0], acc[1][1]);
    q0.z = pk2(acc[2][0], acc[2][1]);  q0.w = pk2(acc[3][0], acc[3][1]);
    q1.x = pk2(acc[4][0], acc[4][1]);  q1.y = pk2(acc[5][0], acc[5][1]);
    q1.z = pk2(acc[6][0], acc[6][1]);  q1.w = pk2(acc[7][0], acc[7][1]);
    q2.x = pk2(pos[0], pos[1]);        q2.y = pk2(pos[2], pos[3]);
    q2.z = pk2(pos[4], pos[5]);        q2.w = pk2(pos[6], pos[7]);
    q3.x = pk2(pos[8], 0.0f);          q3.y = 0u;  q3.z = 0u;  q3.w = 0u;

    _Float16* hr = sH + wv * (32 * KPAD) + lane * KPAD;
    *(uint4*)(hr +  0) = q0;
    *(uint4*)(hr +  8) = q1;
    *(uint4*)(hr + 16) = q2;
    *(uint4*)(hr + 24) = q3;

    __syncthreads();   // sW1 visible to all waves; sH rows visible within wave

    // ---- MLP layer 1 via WMMA (f16 in, f32 acc), swish, layer-2 dot ----
    const int   colbase = lane & 15;
    const int   grp     = lane >> 4;       // 0: lanes 0-15, 1: lanes 16-31
    const float b2v     = b2[0];
    const int   pbase0  = blockIdx.x * BLOCK + wv * 32;

    #pragma unroll
    for (int m = 0; m < 2; ++m) {
        // A operand: row = m*16 + (lane&15); K halves per ISA 16-bit A layout
        const int rowA = m * 16 + colbase;
        const int koff = grp << 3;                                   // 0 or 8
        const _Float16* ap = sH + wv * (32 * KPAD) + rowA * KPAD + koff;
        V16U A;
        A.q[0] = *(const uint4*)(ap);         // K = koff .. koff+7
        A.q[1] = *(const uint4*)(ap + 16);    // K = koff+16 .. koff+23

        float psum[8];
        #pragma unroll
        for (int r = 0; r < 8; ++r) psum[r] = 0.0f;

        #pragma unroll
        for (int n = 0; n < 4; ++n) {
            const int col = n * 16 + colbase;
            const int kb  = grp << 3;                                // 0 or 8
            const _Float16* bp = sW1 + col * KPAD + kb;
            V16U B;
            B.q[0] = *(const uint4*)(bp);
            B.q[1] = *(const uint4*)(bp + 16);

            // C init = b1[col] broadcast (every C element of this lane is column `col`)
            const float b1c = b1[col];
            v8f C;
            #pragma unroll
            for (int r = 0; r < 8; ++r) C[r] = b1c;

            C = __builtin_amdgcn_wmma_f32_16x16x32_f16(
                    false, A.h, false, B.h, (short)0, C, false, false);

            // swish + second-layer weight, accumulated per row.
            // Hardware v_rcp_f32 (not IEEE div expansion) for the sigmoid.
            const float w2c = w2[col];
            #pragma unroll
            for (int r = 0; r < 8; ++r) {
                float v = C[r];
                float s = v * __builtin_amdgcn_rcpf(1.0f + __expf(-v));   // swish
                psum[r] = fmaf(s, w2c, psum[r]);
            }
        }

        // Reduce across the 16 lanes of this column group via ds_swizzle XOR
        // butterfly (masks 1/2/4/8 stay inside each 16-lane group).
        butterfly_step<1>(psum);
        butterfly_step<2>(psum);
        butterfly_step<4>(psum);
        butterfly_step<8>(psum);

        // C row mapping: lanes 0-15 hold M=r, lanes 16-31 hold M=8+r
        if (colbase == 0) {
            const int mbase = pbase0 + m * 16 + (grp << 3);
            #pragma unroll
            for (int r = 0; r < 8; ++r) {
                int po = mbase + r;
                if (po < npts) out[po] = psum[r] + b2v;
            }
        }
    }
}

extern "C" void kernel_launch(void* const* d_in, const int* in_sizes, int n_in,
                              void* d_out, int out_size, void* d_ws, size_t ws_size,
                              hipStream_t stream)
{
    const float* x    = (const float*)d_in[0];
    const float* grid = (const float*)d_in[1];
    const float* w1   = (const float*)d_in[2];
    const float* b1   = (const float*)d_in[3];
    const float* w2   = (const float*)d_in[4];
    const float* b2   = (const float*)d_in[5];
    float* out        = (float*)d_out;

    const int npts   = in_sizes[0] / 3;
    const int blocks = (npts + BLOCK - 1) / BLOCK;
    gridnet3d_kernel<<<blocks, BLOCK, 0, stream>>>(x, grid, w1, b1, w2, b2, out, npts);
}
